// SwitchMoELayer_85710367359163
// MI455X (gfx1250) — matrix operand order repaired and checked by reading, not compiled, not run on hardware
//
#include <hip/hip_runtime.h>
#include <hip/hip_bf16.h>

// ---------------------------------------------------------------------------
// Switch-MoE (top-1) for MI455X / gfx1250, wave32 + WMMA bf16.
//   T=8192 tokens, D=1024, F=4096, E=8 experts.
// Routed FLOPs = 4*T*D*F ~= 137 GFLOP; HBM floor ~= 300 MB -> compute and
// memory are balanced, so we use v_wmma_f32_16x16x32_bf16 (bf16 in, f32 acc)
// with on-the-fly fp32->bf16 weight conversion (v_cvt_pk_bf16_f32 overlaps
// WMMA on the XDL pipe). Per-expert weight slice (33.5 MB fp32) is L2-resident
// (192 MB L2), so the cross-tile W re-reads stay on-chip.
// Workspace: cnt(32B) | toklist(256KB) | x_bf16(16MB) | h_bf16(64MB) ~ 84MB.
// ---------------------------------------------------------------------------

#define T_TOK   8192
#define DM      1024
#define DF      4096
#define NEXP    8

typedef __attribute__((ext_vector_type(16))) __bf16          v16bf;
typedef __attribute__((ext_vector_type(8)))  float           v8f;
typedef __attribute__((ext_vector_type(8)))  unsigned short  us8;

union ABFrag {
    v16bf           v;
    us8             u[2];
    __hip_bfloat16  h[16];
};

__global__ void moe_zero_cnt(int* cnt) {
    if (threadIdx.x < NEXP) cnt[threadIdx.x] = 0;
}

// One wave per token: logits + argmax routing, and x fp32 -> bf16 conversion.
__global__ __launch_bounds__(256)
void moe_router(const float* __restrict__ x,
                const float* __restrict__ Wr,
                const float* __restrict__ br,
                __hip_bfloat16* __restrict__ xb,
                int* __restrict__ cnt,
                int* __restrict__ toklist) {
    const int wave = threadIdx.x >> 5;
    const int lane = threadIdx.x & 31;
    const int t    = blockIdx.x * 8 + wave;

    const float* xr = x + (size_t)t * DM;
    float acc[NEXP];
#pragma unroll
    for (int e = 0; e < NEXP; ++e) acc[e] = 0.0f;

#pragma unroll 4
    for (int i = 0; i < DM / 32; ++i) {
        const int   d = i * 32 + lane;
        const float v = xr[d];
        xb[(size_t)t * DM + d] = __float2bfloat16(v);
#pragma unroll
        for (int e = 0; e < NEXP; ++e)
            acc[e] += v * Wr[e * DM + d];
    }
    // wave32 butterfly reduction
#pragma unroll
    for (int e = 0; e < NEXP; ++e) {
#pragma unroll
        for (int s = 16; s > 0; s >>= 1)
            acc[e] += __shfl_xor(acc[e], s, 32);
    }
    if (lane == 0) {
        int   best = 0;
        float bv   = acc[0] + br[0];
#pragma unroll
        for (int e = 1; e < NEXP; ++e) {
            const float le = acc[e] + br[e];
            if (le > bv) { bv = le; best = e; }   // strict '>' == first-index tie-break
        }
        const int pos = atomicAdd(&cnt[best], 1);
        toklist[best * T_TOK + pos] = t;
    }
}

// Grouped GEMM over one expert's token list.
//   FFN1: h = relu(x_bf16 @ W1[e]^T + b1[e])  (KDIM=DM,  NDIM=DF, store bf16)
//   FFN2: y =       h     @ W2[e]^T + b2[e]   (KDIM=DF,  NDIM=DM, store f32)
// Block = 8 waves, block tile 64(M) x 256(N); wave tile 32x64 (2x4 WMMA frags).
template <int KDIM, bool FFN1>
__global__ __launch_bounds__(256)
void moe_gemm(const __hip_bfloat16* __restrict__ A,   // (T, KDIM) bf16
              const float* __restrict__ W,            // (E, NDIM, KDIM) f32
              const float* __restrict__ bias,         // (E, NDIM)
              const int* __restrict__ cnt,
              const int* __restrict__ toklist,
              __hip_bfloat16* __restrict__ Hout,      // FFN1 out (T, DF)
              float* __restrict__ Yout) {             // FFN2 out (T, DM)
    const int NDIM = FFN1 ? DF : DM;
    const int e    = blockIdx.z;
    const int cntE = cnt[e];
    const int m0   = blockIdx.x * 64;
    if (m0 >= cntE) return;                           // block-uniform: EXEC stays all-1s
    const int n0   = blockIdx.y * 256;

    __shared__ int toks[64];
    if (threadIdx.x < 64) {
        int idx = m0 + (int)threadIdx.x;
        if (idx > cntE - 1) idx = cntE - 1;           // clamp tail (stores are guarded)
        toks[threadIdx.x] = toklist[e * T_TOK + idx];
    }
    __syncthreads();

    const int wave  = threadIdx.x >> 5;
    const int lane  = threadIdx.x & 31;
    const int wm    = wave >> 2;      // 0..1 -> M offset 0/32
    const int wn    = wave & 3;       // 0..3 -> N offset 0/64/128/192
    const int mlo   = lane & 15;      // M (A) / N (B,C) index within fragment
    const int khalf = lane >> 4;      // K half selector per ISA 16-bit layouts

    const __hip_bfloat16* arow[2];
#pragma unroll
    for (int mf = 0; mf < 2; ++mf)
        arow[mf] = A + (size_t)toks[wm * 32 + mf * 16 + mlo] * KDIM;

    const float* wrow[4];
#pragma unroll
    for (int nf = 0; nf < 4; ++nf) {
        const int ncol = n0 + wn * 64 + nf * 16 + mlo;
        wrow[nf] = W + ((size_t)e * NDIM + ncol) * KDIM;
    }

    v8f acc[2][4];
#pragma unroll
    for (int mf = 0; mf < 2; ++mf)
#pragma unroll
        for (int nf = 0; nf < 4; ++nf)
#pragma unroll
            for (int i = 0; i < 8; ++i) acc[mf][nf][i] = 0.0f;

    for (int k0 = 0; k0 < KDIM; k0 += 32) {
        // A fragment: contiguous K order (k = 16*khalf + s), matching B -> 16 contiguous bf16/lane
        ABFrag a[2];
#pragma unroll
        for (int mf = 0; mf < 2; ++mf) {
            a[mf].u[0] = *(const us8*)(arow[mf] + k0 + khalf * 16);
            a[mf].u[1] = *(const us8*)(arow[mf] + k0 + khalf * 16 + 8);
        }
        // B fragment: column N = weight row (contiguous in K); load f32, cvt->bf16.
        // Unconditional prefetch 2 K-steps ahead: speculative, silently dropped
        // if past the mapped range (ISA 10.5), so no in-loop branch needed.
        ABFrag b[4];
#pragma unroll
        for (int nf = 0; nf < 4; ++nf) {
            const float* p = wrow[nf] + k0 + khalf * 16;
            __builtin_prefetch(p + 64, 0, 1);
            const float4 f0 = *(const float4*)(p + 0);
            const float4 f1 = *(const float4*)(p + 4);
            const float4 f2 = *(const float4*)(p + 8);
            const float4 f3 = *(const float4*)(p + 12);
            const float  fl[16] = {f0.x, f0.y, f0.z, f0.w, f1.x, f1.y, f1.z, f1.w,
                                   f2.x, f2.y, f2.z, f2.w, f3.x, f3.y, f3.z, f3.w};
#pragma unroll
            for (int i = 0; i < 16; ++i) b[nf].h[i] = __float2bfloat16(fl[i]);
        }
#pragma unroll
        for (int mf = 0; mf < 2; ++mf)
#pragma unroll
            for (int nf = 0; nf < 4; ++nf)
                acc[mf][nf] = __builtin_amdgcn_wmma_f32_16x16x32_bf16(
                    false, a[mf].v, false, b[nf].v,
                    (short)0, acc[mf][nf], false, false);
    }

    // Epilogue. C layout: VGPR r, lane = 16*khalf + mlo -> M = r + 8*khalf, N = mlo.
#pragma unroll
    for (int nf = 0; nf < 4; ++nf) {
        const int   ncol = n0 + wn * 64 + nf * 16 + mlo;
        const float bv   = bias[e * NDIM + ncol];
#pragma unroll
        for (int mf = 0; mf < 2; ++mf) {
#pragma unroll
            for (int r = 0; r < 8; ++r) {
                const int rowLocal = wm * 32 + mf * 16 + khalf * 8 + r;
                if (m0 + rowLocal < cntE) {
                    float v = acc[mf][nf][r] + bv;
                    const int tok = toks[rowLocal];
                    if (FFN1) {
                        v = v > 0.0f ? v : 0.0f;
                        Hout[(size_t)tok * DF + ncol] = __float2bfloat16(v);
                    } else {
                        Yout[(size_t)tok * DM + ncol] = v;
                    }
                }
            }
        }
    }
}

extern "C" void kernel_launch(void* const* d_in, const int* in_sizes, int n_in,
                              void* d_out, int out_size, void* d_ws, size_t ws_size,
                              hipStream_t stream) {
    (void)in_sizes; (void)n_in; (void)out_size; (void)ws_size;

    const float* x  = (const float*)d_in[0];
    const float* Wr = (const float*)d_in[1];
    const float* br = (const float*)d_in[2];
    const float* W1 = (const float*)d_in[3];
    const float* b1 = (const float*)d_in[4];
    const float* W2 = (const float*)d_in[5];
    const float* b2 = (const float*)d_in[6];
    float*       out = (float*)d_out;

    // Workspace carve-up (~84.2 MB total)
    char* ws = (char*)d_ws;
    int*  cnt     = (int*)ws;                                        // 8 ints
    int*  toklist = (int*)(ws + 256);                                // E*T ints (256 KB)
    __hip_bfloat16* xb = (__hip_bfloat16*)(ws + 256 + (size_t)NEXP * T_TOK * 4);   // 16 MB
    __hip_bfloat16* hb = (__hip_bfloat16*)((char*)xb + (size_t)T_TOK * DM * 2);    // 64 MB

    moe_zero_cnt<<<1, 32, 0, stream>>>(cnt);
    moe_router<<<T_TOK / 8, 256, 0, stream>>>(x, Wr, br, xb, cnt, toklist);
    moe_gemm<DM, true ><<<dim3(T_TOK / 64, DF / 256, NEXP), 256, 0, stream>>>(
        xb, W1, b1, cnt, toklist, hb, nullptr);
    moe_gemm<DF, false><<<dim3(T_TOK / 64, DM / 256, NEXP), 256, 0, stream>>>(
        hb, W2, b2, cnt, toklist, nullptr, out);
}
